// ScaledDotProdAttV1_49022756716708
// MI455X (gfx1250) — matrix operand order repaired and checked by reading, not compiled, not run on hardware
//
#include <hip/hip_runtime.h>
#include <hip/hip_bf16.h>

#define BB   4
#define TT   2048
#define DD   1024
#define HH   16
#define DKK  64
#define DVV  64
#define TQ_TILES (TT / 16)

typedef __attribute__((ext_vector_type(16))) _Float16 v16h;
typedef __attribute__((ext_vector_type(8)))  float    v8f;
typedef __attribute__((ext_vector_type(4)))  unsigned int v4u;
typedef __attribute__((ext_vector_type(4)))  int      v4i;

#define GLOBAL_AS __attribute__((address_space(1)))
#define LDS_AS    __attribute__((address_space(3)))

#if defined(__has_builtin)
#if __has_builtin(__builtin_amdgcn_global_load_async_to_lds_b128)
#define HAVE_ASYNC_LDS 1
#endif
#endif

enum { OUT_F32 = 0, OUT_QK = 1, OUT_VT = 2 };

// ---------------------------------------------------------------------------
// Generic  out = A @ W^T + bias  GEMM.  A: MxK f32 row-major, W: NxK f32
// row-major.  One wave computes a 16x64 strip: the A fragment is loaded once
// per K-step and reused across 4 WMMAs (4x less A traffic).
// ---------------------------------------------------------------------------
__global__ __launch_bounds__(256)
void gemm_wmma_f16(const float* __restrict__ A, int lda,
                   const float* __restrict__ W, int ldw,
                   const float* __restrict__ bias,
                   float* __restrict__ outF,
                   _Float16* __restrict__ outH,
                   int M, int N, int K, int mode)
{
    const int wave = threadIdx.x >> 5;
    const int lane = threadIdx.x & 31;
    const int n0 = (blockIdx.x * 8 + wave) * 64;     // 4 N-tiles per wave
    const int m0 = blockIdx.y * 16;
    if (n0 >= N || m0 >= M) return;                  // wave-uniform

    const int arow  = lane & 15;                     // A: row within tile
    const int sel   = lane >> 4;                     // half-wave selector
    const int akoff = sel * 16;                      // A: K sub-offset {0,16}
    const int bcol  = lane & 15;                     // B: column within tile
    const int bkoff = sel * 16;                      // B: K sub-offset {0,16}

    const float* aptr = A + (size_t)(m0 + arow) * lda + akoff;
    const float* wptr = W + (size_t)(n0 + bcol) * ldw + bkoff;

    v8f c[4] = {};
    for (int k0 = 0; k0 < K; k0 += 32) {
        v16h af;
#pragma unroll
        for (int i = 0; i < 8; ++i) {                // A 16x32 frag: 16-run
            af[i]     = (_Float16)aptr[k0 + i];
            af[8 + i] = (_Float16)aptr[k0 + 8 + i];
        }
#pragma unroll
        for (int j = 0; j < 4; ++j) {
            const float* wp = wptr + (size_t)j * 16 * ldw + k0;
            v16h bf;
#pragma unroll
            for (int i = 0; i < 16; ++i)             // B 32x16 frag: 16-run
                bf[i] = (_Float16)wp[i];
            c[j] = __builtin_amdgcn_wmma_f32_16x16x32_f16(
                       false, af, false, bf, (short)0, c[j], false, false);
        }
    }

    const int rb = sel * 8;                          // C/D rows rb..rb+7
#pragma unroll
    for (int j = 0; j < 4; ++j) {
        const int ncol = n0 + j * 16 + bcol;
        const float bval = bias[ncol];
        if (mode == OUT_F32) {
#pragma unroll
            for (int r = 0; r < 8; ++r)
                outF[(size_t)(m0 + rb + r) * N + ncol] = c[j][r] + bval;
        } else if (mode == OUT_QK) {                 // f16 [b][h][t][dk]
            const int h = ncol / DKK, dk = ncol % DKK;
#pragma unroll
            for (int r = 0; r < 8; ++r) {
                const int m = m0 + rb + r, b = m / TT, t = m % TT;
                outH[(((size_t)b * HH + h) * TT + t) * DKK + dk] =
                    (_Float16)(c[j][r] + bval);
            }
        } else {                                     // f16 [b][h][dv][t]
            const int h = ncol / DVV, dv = ncol % DVV;
#pragma unroll
            for (int r = 0; r < 8; ++r) {
                const int m = m0 + rb + r, b = m / TT, t = m % TT;
                outH[(((size_t)b * HH + h) * DVV + dv) * TT + t] =
                    (_Float16)(c[j][r] + bval);
            }
        }
    }
}

// ---------------------------------------------------------------------------
// Flash-attention: block = 4 waves sharing one (b,h), 4 consecutive 16-row
// query tiles.  Per 32-key block: K (4KB, contiguous) and V^T (4KB) are
// staged into LDS with async global->LDS copies (ASYNCcnt path), then each
// wave runs 4 S-WMMAs + masked online softmax + LDS P-transpose + 4 PV-WMMAs.
// ---------------------------------------------------------------------------
__global__ __launch_bounds__(128)
void attn_wmma_f16(const _Float16* __restrict__ qh,   // [b][h][t][dk]
                   const _Float16* __restrict__ kh,   // [b][h][t][dk]
                   const _Float16* __restrict__ vt,   // [b][h][dv][t]
                   const int* __restrict__ mask,      // [b][t][t]
                   float* __restrict__ xf)            // [b][t][h*dv]
{
    __shared__ _Float16 kTile[32][64];                // 4KB: [key][dk]
    __shared__ _Float16 vTile[64][32];                // 4KB: [dv][key]
    __shared__ _Float16 pTile[4][16][40];             // P transpose, per wave

    const int tid  = threadIdx.x;
    const int wave = tid >> 5;
    const int lane = tid & 31;

    const int b   = blockIdx.x / (HH * (TQ_TILES / 4));
    const int rem = blockIdx.x % (HH * (TQ_TILES / 4));
    const int h   = rem / (TQ_TILES / 4);
    const int tq0 = ((rem % (TQ_TILES / 4)) * 4 + wave) * 16;

    const size_t bh = (size_t)b * HH + h;
    const _Float16* qbase = qh + (bh * TT + tq0) * DKK;
    const _Float16* kbase = kh + bh * TT * DKK;
    const _Float16* vbase = vt + bh * DVV * TT;

    const int row16 = lane & 15;
    const int sel   = lane >> 4;
    const int rb    = sel * 8;

    // Q A-fragments for dk chunks [0,32) and [32,64), loaded once.
    v16h qa0, qa1;
    {
        const _Float16* p = qbase + (size_t)row16 * DKK + sel * 16;
#pragma unroll
        for (int i = 0; i < 8; ++i) {
            qa0[i]     = p[i];       qa0[8 + i] = p[8 + i];
            qa1[i]     = p[32 + i];  qa1[8 + i] = p[40 + i];
        }
    }

    v8f oacc[4] = {};
    float mrun[8], lrun[8];
#pragma unroll
    for (int r = 0; r < 8; ++r) { mrun[r] = -3.0e38f; lrun[r] = 0.0f; }

    const float scale = 0.125f;                       // 1/sqrt(DK)
    const int* mrow = mask + ((size_t)b * TT + tq0 + rb) * TT;

    for (int tk0 = 0; tk0 < TT; tk0 += 32) {
        __syncthreads();                              // readers of prev block done

        // ---- stage K(32x64) and V^T(64x32) f16 blocks into LDS
        {
            const char* kblk = (const char*)(kbase + (size_t)tk0 * DKK); // 4KB contiguous
            char* kdst = (char*)&kTile[0][0];
            char* vdst = (char*)&vTile[0][0];
#pragma unroll
            for (int it = 0; it < 2; ++it) {
                const int c  = tid + it * 128;        // 16B chunk id, 0..255
                const int dv = c >> 2, part = c & 3;  // V row / 16B part
                const char* vsrc =
                    (const char*)(vbase + (size_t)dv * TT + tk0) + part * 16;
#if defined(HAVE_ASYNC_LDS)
                __builtin_amdgcn_global_load_async_to_lds_b128(
                    (GLOBAL_AS v4i*)(kblk + c * 16),
                    (LDS_AS v4i*)(kdst + c * 16), 0, 0);
                __builtin_amdgcn_global_load_async_to_lds_b128(
                    (GLOBAL_AS v4i*)vsrc,
                    (LDS_AS v4i*)(vdst + c * 16), 0, 0);
#else
                *(v4u*)(kdst + c * 16) = *(const v4u*)(kblk + c * 16);
                *(v4u*)(vdst + c * 16) = *(const v4u*)vsrc;
#endif
            }
#if defined(HAVE_ASYNC_LDS)
            asm volatile("s_wait_asynccnt 0x0" ::: "memory");
#endif
        }
        if (tk0 + 32 < TT)                            // pull next K block
            __builtin_prefetch(kbase + (size_t)(tk0 + 32) * DKK + tid * 16, 0, 0);
        __syncthreads();                              // staged tiles visible

        // ---- S = Q K^T, 16x32 block as two 16x16 WMMA tiles (K from LDS)
        v8f s0 = {}, s1 = {};
        {
            const _Float16* kp0 = &kTile[row16][sel * 16];
            const _Float16* kp1 = &kTile[16 + row16][sel * 16];
            v16h k00, k01, k10, k11;
#pragma unroll
            for (int i = 0; i < 16; ++i) {
                k00[i] = kp0[i];       k01[i] = kp0[32 + i];
                k10[i] = kp1[i];       k11[i] = kp1[32 + i];
            }
            s0 = __builtin_amdgcn_wmma_f32_16x16x32_f16(false, qa0, false, k00,
                                                        (short)0, s0, false, false);
            s0 = __builtin_amdgcn_wmma_f32_16x16x32_f16(false, qa1, false, k01,
                                                        (short)0, s0, false, false);
            s1 = __builtin_amdgcn_wmma_f32_16x16x32_f16(false, qa0, false, k10,
                                                        (short)0, s1, false, false);
            s1 = __builtin_amdgcn_wmma_f32_16x16x32_f16(false, qa1, false, k11,
                                                        (short)0, s1, false, false);
        }

        // ---- mask bits (1 = masked out)
        unsigned mb0 = 0, mb1 = 0;
#pragma unroll
        for (int r = 0; r < 8; ++r) {
            const int* mp = mrow + (size_t)r * TT + tk0;
            if (mp[row16] == 0)      mb0 |= 1u << r;
            if (mp[16 + row16] == 0) mb1 |= 1u << r;
        }

        // ---- online softmax per row (rows live in 16-lane halves)
        float p0[8], p1[8], alpha[8];
#pragma unroll
        for (int r = 0; r < 8; ++r) {
            float v0 = ((mb0 >> r) & 1) ? -3.0e38f : s0[r] * scale;
            float v1 = ((mb1 >> r) & 1) ? -3.0e38f : s1[r] * scale;
            float mx = fmaxf(v0, v1);
            mx = fmaxf(mx, __shfl_xor(mx, 1, 16));
            mx = fmaxf(mx, __shfl_xor(mx, 2, 16));
            mx = fmaxf(mx, __shfl_xor(mx, 4, 16));
            mx = fmaxf(mx, __shfl_xor(mx, 8, 16));
            const float mnew = fmaxf(mrun[r], mx);
            const float a  = __expf(mrun[r] - mnew);
            const float e0 = ((mb0 >> r) & 1) ? 0.0f : __expf(v0 - mnew);
            const float e1 = ((mb1 >> r) & 1) ? 0.0f : __expf(v1 - mnew);
            float rs = e0 + e1;
            rs += __shfl_xor(rs, 1, 16);
            rs += __shfl_xor(rs, 2, 16);
            rs += __shfl_xor(rs, 4, 16);
            rs += __shfl_xor(rs, 8, 16);
            lrun[r] = lrun[r] * a + rs;
            mrun[r] = mnew;
            alpha[r] = a;
            p0[r] = e0;
            p1[r] = e1;
        }
#pragma unroll
        for (int r = 0; r < 8; ++r) {
            oacc[0][r] *= alpha[r]; oacc[1][r] *= alpha[r];
            oacc[2][r] *= alpha[r]; oacc[3][r] *= alpha[r];
        }

        // ---- P: C-layout -> A-layout via per-wave LDS tile
#pragma unroll
        for (int r = 0; r < 8; ++r) {
            pTile[wave][rb + r][row16]      = (_Float16)p0[r];
            pTile[wave][rb + r][16 + row16] = (_Float16)p1[r];
        }
        asm volatile("s_wait_dscnt 0x0" ::: "memory");  // wave-local RAW fence
        v16h pa;
        {
            const _Float16* pp = &pTile[wave][row16][sel * 16];
#pragma unroll
            for (int i = 0; i < 8; ++i) { pa[i] = pp[i]; pa[8 + i] = pp[8 + i]; }
        }

        // ---- O += P @ V  (4 N-tiles of 16 over DV=64, V from LDS)
#pragma unroll
        for (int jn = 0; jn < 4; ++jn) {
            const _Float16* vp = &vTile[jn * 16 + row16][sel * 16];
            v16h vb;
#pragma unroll
            for (int i = 0; i < 16; ++i) vb[i] = vp[i];
            oacc[jn] = __builtin_amdgcn_wmma_f32_16x16x32_f16(
                false, pa, false, vb, (short)0, oacc[jn], false, false);
        }
    }

    // ---- normalize and store x[b][t][h*DV]
#pragma unroll
    for (int r = 0; r < 8; ++r) {
        const float inv = (lrun[r] > 0.0f) ? (1.0f / lrun[r]) : 0.0f;
        float* xp = xf + ((size_t)b * TT + tq0 + rb + r) * (HH * DVV) + h * DVV;
        xp[ 0 + row16] = oacc[0][r] * inv;
        xp[16 + row16] = oacc[1][r] * inv;
        xp[32 + row16] = oacc[2][r] * inv;
        xp[48 + row16] = oacc[3][r] * inv;
    }
}

// ---------------------------------------------------------------------------
extern "C" void kernel_launch(void* const* d_in, const int* in_sizes, int n_in,
                              void* d_out, int out_size, void* d_ws, size_t ws_size,
                              hipStream_t stream) {
    (void)in_sizes; (void)n_in; (void)out_size; (void)ws_size;

    const float* query = (const float*)d_in[0];
    const float* key   = (const float*)d_in[1];
    const float* value = (const float*)d_in[2];
    const int*   mask  = (const int*)d_in[3];
    const float* Wq    = (const float*)d_in[4];
    const float* bq    = (const float*)d_in[5];
    const float* Wk    = (const float*)d_in[6];
    const float* bk    = (const float*)d_in[7];
    const float* Wv    = (const float*)d_in[8];
    const float* bv    = (const float*)d_in[9];
    const float* Wo    = (const float*)d_in[10];
    const float* bo    = (const float*)d_in[11];
    float* out = (float*)d_out;

    // workspace: qh (16MB f16) | kh (16MB) | vt (16MB) | xf (32MB f32)
    const size_t nQK = (size_t)BB * HH * TT * DKK;
    _Float16* qh = (_Float16*)d_ws;
    _Float16* kh = qh + nQK;
    _Float16* vt = kh + nQK;
    float*    xf = (float*)(vt + nQK);

    const int M = BB * TT;                 // 8192
    dim3 gblk(256);
    dim3 ggrid(DD / 512, M / 16);          // 8 waves x 64 cols = 512 cols/block

    gemm_wmma_f16<<<ggrid, gblk, 0, stream>>>(query, DD, Wq, DD, bq,
                                              nullptr, qh, M, HH * DKK, DD, OUT_QK);
    gemm_wmma_f16<<<ggrid, gblk, 0, stream>>>(key,   DD, Wk, DD, bk,
                                              nullptr, kh, M, HH * DKK, DD, OUT_QK);
    gemm_wmma_f16<<<ggrid, gblk, 0, stream>>>(value, DD, Wv, DD, bv,
                                              nullptr, vt, M, HH * DVV, DD, OUT_VT);

    attn_wmma_f16<<<dim3(BB * HH * (TQ_TILES / 4)), dim3(128), 0, stream>>>(
        qh, kh, vt, mask, xf);

    gemm_wmma_f16<<<ggrid, gblk, 0, stream>>>(xf, HH * DVV, Wo, HH * DVV, bo,
                                              out, nullptr, M, DD, HH * DVV, OUT_F32);
}
